// Attention_17154099380987
// MI455X (gfx1250) — matrix operand order repaired and checked by reading, not compiled, not run on hardware
//
#include <hip/hip_runtime.h>
#include <hip/hip_bf16.h>
#include <stdint.h>

// MHA forward for B=4, S=2048, D=1024, H=16, HD=64 on gfx1250 (wave32, WMMA).
// Pipeline: 3x f16-WMMA GEMM projections -> flash attention (f16 WMMA, f32
// online softmax, async global->LDS staging) -> f16-WMMA output projection.

typedef __attribute__((ext_vector_type(16))) _Float16 v16h;
typedef __attribute__((ext_vector_type(8)))  _Float16 v8h;
typedef __attribute__((ext_vector_type(8)))  float    v8f;

#define DD  1024
#define SS  2048
#define HH  16
#define HDD 64

union HV { v16h v; v8h h[2]; };

// A fragment (16x32 f16): base points at (row0, k0), row-major stride ld (halfs).
// Per ISA: lanes 0-15 rows 0-15 K{0-7,16-23}; lanes 16-31 rows 0-15 K{8-15,24-31}.
static __device__ inline v16h load_a_frag(const _Float16* base, int ld, int lane) {
  int row  = lane & 15;
  int koff = (lane < 16) ? 0 : 8;
  HV u;
  u.h[0] = *(const v8h*)(base + (size_t)row * ld + koff);
  u.h[1] = *(const v8h*)(base + (size_t)row * ld + 16 + koff);
  return u.v;
}

// B fragment (32x16 f16) read from B^T stored row-major [n][k] with stride ld.
// Lane n = lane&15; lanes 0-15 K{0-7,16-23}, lanes 16-31 K{8-15,24-31}.
static __device__ inline v16h load_bt_frag(const _Float16* base, int ld, int lane) {
  int n    = lane & 15;
  int koff = (lane < 16) ? 0 : 8;
  HV u;
  u.h[0] = *(const v8h*)(base + (size_t)n * ld + koff);
  u.h[1] = *(const v8h*)(base + (size_t)n * ld + 16 + koff);
  return u.v;
}

static __device__ inline v8f wmma_f16(v16h a, v16h b, v8f c) {
  return __builtin_amdgcn_wmma_f32_16x16x32_f16(false, a, false, b, (short)0, c,
                                                false, false);
}

// Async global->LDS copy of 16 bytes (ASYNCcnt-tracked). lds_addr is the low
// 32 bits of the generic pointer to the LDS destination; goff is a byte offset
// added to the 64-bit SGPR base (GVS addressing).
static __device__ inline void async_copy_b128(unsigned lds_addr, unsigned goff,
                                              unsigned long long base) {
  asm volatile("global_load_async_to_lds_b128 %0, %1, %2"
               :: "v"(lds_addr), "v"(goff), "s"(base) : "memory");
}
static __device__ inline void wait_async0() {
  asm volatile("s_wait_asynccnt 0x0" ::: "memory");
}

// ---------------------------------------------------------------------------
// GEMM: Out[M=8192, N=1024] = X[M,1024] @ W[1024,1024] (+bias) (+lev)
// Block tile 128x64, 8 waves (4x2), wave tile 32x32 (2x2 WMMA frags), K step 64.
// ---------------------------------------------------------------------------
template <bool IN_F16, bool ADD_LEV, bool OUT_HEADS>
__global__ __launch_bounds__(256) void gemm_wmma(
    const void* __restrict__ Xv, const float* __restrict__ W,
    const float* __restrict__ bias, const float* __restrict__ lev,
    void* __restrict__ Out) {
  __shared__ __align__(16) _Float16 sA[128 * 72];  // [row][k], stride 72 halfs
  __shared__ __align__(16) _Float16 sB[64 * 72];   // B^T: [n][k], stride 72

  const int t = threadIdx.x;
  const int wave = t >> 5, lane = t & 31;
  const int wm = wave >> 1, wn = wave & 1;
  const int rowBase = blockIdx.y * 128;
  const int nBase   = blockIdx.x * 64;

  v8f zero = {0.f, 0.f, 0.f, 0.f, 0.f, 0.f, 0.f, 0.f};
  v8f c00 = zero, c01 = zero, c10 = zero, c11 = zero;

  for (int kk = 0; kk < DD; kk += 64) {
    // Stage A tile (128x64 -> f16)
    if (IN_F16) {
      const _Float16* X = (const _Float16*)Xv;
#pragma unroll
      for (int i = 0; i < 4; ++i) {
        int c = t + 256 * i;                 // 1024 chunks of 8 halfs
        int row = c >> 3, col8 = (c & 7) * 8;
        *(v8h*)(sA + row * 72 + col8) =
            *(const v8h*)(X + (size_t)(rowBase + row) * DD + kk + col8);
      }
    } else {
      const float* X = (const float*)Xv;
#pragma unroll
      for (int i = 0; i < 8; ++i) {
        int c = t + 256 * i;                 // 2048 chunks of float4
        int row = c >> 4, col4 = (c & 15) * 4;
        float4 f = *(const float4*)(X + (size_t)(rowBase + row) * DD + kk + col4);
        _Float16* d = sA + row * 72 + col4;
        d[0] = (_Float16)f.x; d[1] = (_Float16)f.y;
        d[2] = (_Float16)f.z; d[3] = (_Float16)f.w;
      }
    }
    // Stage B tile transposed: sB[n][k] = W[kk+k][nBase+n]
#pragma unroll
    for (int i = 0; i < 4; ++i) {
      int c = t + 256 * i;                   // 1024 chunks: 64 k-rows x 16 n4
      int krow = c >> 4, n4 = (c & 15) * 4;
      float4 f = *(const float4*)(W + (size_t)(kk + krow) * DD + nBase + n4);
      sB[(n4 + 0) * 72 + krow] = (_Float16)f.x;
      sB[(n4 + 1) * 72 + krow] = (_Float16)f.y;
      sB[(n4 + 2) * 72 + krow] = (_Float16)f.z;
      sB[(n4 + 3) * 72 + krow] = (_Float16)f.w;
    }
    __syncthreads();

#pragma unroll
    for (int ks = 0; ks < 2; ++ks) {
      const int kB = ks * 32;
      v16h a0 = load_a_frag(sA + (wm * 32) * 72 + kB, 72, lane);
      v16h a1 = load_a_frag(sA + (wm * 32 + 16) * 72 + kB, 72, lane);
      v16h b0 = load_bt_frag(sB + (wn * 32) * 72 + kB, 72, lane);
      v16h b1 = load_bt_frag(sB + (wn * 32 + 16) * 72 + kB, 72, lane);
      c00 = wmma_f16(a0, b0, c00);
      c01 = wmma_f16(a0, b1, c01);
      c10 = wmma_f16(a1, b0, c10);
      c11 = wmma_f16(a1, b1, c11);
    }
    __syncthreads();
  }

  // Epilogue: C frag layout: VGPR r -> M = r + 8*(lane>=16), N = lane&15.
  const int m2 = 8 * (lane >> 4);
  const int nn = lane & 15;
  v8f* accs[4] = {&c00, &c01, &c10, &c11};
#pragma unroll
  for (int f = 0; f < 4; ++f) {
    int mOff = (f >> 1) * 16, nOff = (f & 1) * 16;
    int nl = nBase + wn * 32 + nOff + nn;
    float bvv = bias[nl];
    v8f cc = *accs[f];
#pragma unroll
    for (int r = 0; r < 8; ++r) {
      int ml = rowBase + wm * 32 + mOff + r + m2;
      float val = cc[r] + bvv;
      if (ADD_LEV) val += lev[(size_t)ml * DD + nl];
      if (OUT_HEADS) {
        int b = ml >> 11, s = ml & (SS - 1);
        int h = nl >> 6,  hd = nl & (HDD - 1);
        ((_Float16*)Out)[(((size_t)b * HH + h) * SS + s) * HDD + hd] =
            (_Float16)val;
      } else {
        ((float*)Out)[(size_t)ml * DD + nl] = val;
      }
    }
  }
}

// ---------------------------------------------------------------------------
// Flash attention: one block per (b*h, 128 query rows); wave owns 16 q-rows.
// Key tile 64 per iteration; K tile staged via async global->LDS (ASYNCcnt),
// V tile transposed via LDS scatter. Online softmax in f32 (C-fragment row
// groups live in 16-lane halves -> shfl_xor masks 1,2,4,8 reduce per half).
// ---------------------------------------------------------------------------
__global__ __launch_bounds__(256) void attn_wmma(
    const _Float16* __restrict__ qh, const _Float16* __restrict__ kh,
    const _Float16* __restrict__ vh, _Float16* __restrict__ ctx) {
  __shared__ __align__(16) _Float16 sK[64 * 72];       // [key][d], stride 72
  __shared__ __align__(16) _Float16 sV[64 * 72];       // V^T: [d][key], stride 72
  __shared__ __align__(16) _Float16 sP[8 * 16 * 72];   // per-wave P scratch

  const int t = threadIdx.x;
  const int wave = t >> 5, lane = t & 31;
  const int bh = blockIdx.y;
  const size_t headOff = (size_t)bh * SS * HDD;
  const _Float16* qb = qh + headOff;
  const _Float16* kb = kh + headOff;
  const _Float16* vb = vh + headOff;
  const int q0 = blockIdx.x * 128 + wave * 16;

  // Q fragments (16 rows x 64 d) loaded once straight from global (rows 128B).
  v16h aq[2];
#pragma unroll
  for (int ks = 0; ks < 2; ++ks)
    aq[ks] = load_a_frag(qb + (size_t)q0 * HDD + ks * 32, HDD, lane);

  v8f zero = {0.f, 0.f, 0.f, 0.f, 0.f, 0.f, 0.f, 0.f};
  v8f acc[4] = {zero, zero, zero, zero};
  float mi[8], li[8];
#pragma unroll
  for (int r = 0; r < 8; ++r) { mi[r] = -3.0e38f; li[r] = 0.f; }

  const float scale = 0.125f;  // 1/sqrt(64)
  const int m2 = 8 * (lane >> 4);
  const int nn = lane & 15;
  _Float16* pw = sP + wave * 16 * 72;
  const unsigned long long kbase = (unsigned long long)(uintptr_t)kb;

  for (int kt = 0; kt < SS; kt += 64) {
    // Stage 64x64 K tile via async global->LDS b128 (2 chunks per thread).
#pragma unroll
    for (int i = 0; i < 2; ++i) {
      int c = t + 256 * i;                   // 512 chunks of 8 halfs
      int krow = c >> 3, c8 = (c & 7) * 8;
      unsigned lds_addr =
          (unsigned)(uintptr_t)(const void*)(sK + krow * 72 + c8);
      unsigned goff = (unsigned)(((kt + krow) * HDD + c8) * sizeof(_Float16));
      async_copy_b128(lds_addr, goff, kbase);
    }
    // Stage V tile transposed (normal loads + LDS scatter).
#pragma unroll
    for (int i = 0; i < 2; ++i) {
      int c = t + 256 * i;
      int krow = c >> 3, c8 = (c & 7) * 8;
      v8h vd = *(const v8h*)(vb + (size_t)(kt + krow) * HDD + c8);
#pragma unroll
      for (int j = 0; j < 8; ++j) sV[(c8 + j) * 72 + krow] = vd[j];
      if (kt + 64 < SS) {
        __builtin_prefetch(kb + (size_t)(kt + 64 + krow) * HDD + c8, 0, 1);
        __builtin_prefetch(vb + (size_t)(kt + 64 + krow) * HDD + c8, 0, 1);
      }
    }
    wait_async0();
    __syncthreads();

    // Scores S = q @ k^T (16x64): 4 N-fragments, K-dim = HD = 64 -> 2 k-steps.
    v8f s[4] = {zero, zero, zero, zero};
#pragma unroll
    for (int ks = 0; ks < 2; ++ks) {
#pragma unroll
      for (int nf = 0; nf < 4; ++nf) {
        v16h bk = load_bt_frag(sK + (nf * 16) * 72 + ks * 32, 72, lane);
        s[nf] = wmma_f16(aq[ks], bk, s[nf]);
      }
    }
#pragma unroll
    for (int nf = 0; nf < 4; ++nf) s[nf] *= scale;

    // Online softmax per row (row m = r + m2; replicated across the 16 lanes
    // of this wave-half).
#pragma unroll
    for (int r = 0; r < 8; ++r) {
      float x = fmaxf(fmaxf(s[0][r], s[1][r]), fmaxf(s[2][r], s[3][r]));
      x = fmaxf(x, __shfl_xor(x, 1, 32));
      x = fmaxf(x, __shfl_xor(x, 2, 32));
      x = fmaxf(x, __shfl_xor(x, 4, 32));
      x = fmaxf(x, __shfl_xor(x, 8, 32));
      float mnew  = fmaxf(mi[r], x);
      float alpha = __expf(mi[r] - mnew);
      mi[r] = mnew;
      float rs = 0.f;
#pragma unroll
      for (int nf = 0; nf < 4; ++nf) {
        float e = __expf(s[nf][r] - mnew);
        s[nf][r] = e;
        rs += e;
      }
      rs += __shfl_xor(rs, 1, 32);
      rs += __shfl_xor(rs, 2, 32);
      rs += __shfl_xor(rs, 4, 32);
      rs += __shfl_xor(rs, 8, 32);
      li[r] = li[r] * alpha + rs;
#pragma unroll
      for (int j = 0; j < 4; ++j) acc[j][r] *= alpha;
    }

    // P: C layout -> LDS f16 -> reload as A fragments (same wave only).
#pragma unroll
    for (int r = 0; r < 8; ++r) {
#pragma unroll
      for (int nf = 0; nf < 4; ++nf)
        pw[(m2 + r) * 72 + nf * 16 + nn] = (_Float16)s[nf][r];
    }
    asm volatile("s_wait_dscnt 0x0" ::: "memory");

    // ctx += P(16x64) @ V(64x64): 2 k-steps x 4 N-fragments.
#pragma unroll
    for (int kp = 0; kp < 2; ++kp) {
      v16h ap = load_a_frag(pw + kp * 32, 72, lane);
#pragma unroll
      for (int j = 0; j < 4; ++j) {
        v16h bvf = load_bt_frag(sV + (j * 16) * 72 + kp * 32, 72, lane);
        acc[j] = wmma_f16(ap, bvf, acc[j]);
      }
    }
    __syncthreads();
  }

  // Normalize and store ctx (merged heads, f16, [B,S,D]).
  const int b = bh >> 4, h = bh & (HH - 1);
#pragma unroll
  for (int j = 0; j < 4; ++j) {
    int col = h * HDD + j * 16 + nn;
#pragma unroll
    for (int r = 0; r < 8; ++r) {
      int s2 = q0 + m2 + r;
      ctx[((size_t)b * SS + s2) * DD + col] = (_Float16)(acc[j][r] / li[r]);
    }
  }
}

// ---------------------------------------------------------------------------
extern "C" void kernel_launch(void* const* d_in, const int* in_sizes, int n_in,
                              void* d_out, int out_size, void* d_ws,
                              size_t ws_size, hipStream_t stream) {
  const float* Q    = (const float*)d_in[0];
  const float* Qlev = (const float*)d_in[1];
  const float* K    = (const float*)d_in[2];
  const float* Klev = (const float*)d_in[3];
  const float* V    = (const float*)d_in[4];
  const float* Vlev = (const float*)d_in[5];
  const float* Wq   = (const float*)d_in[6];
  const float* bq   = (const float*)d_in[7];
  const float* Wk   = (const float*)d_in[8];
  const float* bk   = (const float*)d_in[9];
  const float* Wv   = (const float*)d_in[10];
  const float* bv   = (const float*)d_in[11];
  const float* Wo   = (const float*)d_in[12];
  const float* bo   = (const float*)d_in[13];

  const size_t HEAD_ELEMS = (size_t)4 * HH * SS * HDD;  // 8,388,608 halfs
  _Float16* qh  = (_Float16*)d_ws;
  _Float16* kh  = qh + HEAD_ELEMS;
  _Float16* vh  = kh + HEAD_ELEMS;
  _Float16* ctx = vh + HEAD_ELEMS;

  dim3 gproj(16, 64);   // N/64, M/128  (M = B*S = 8192, N = D = 1024)
  dim3 gattn(16, 64);   // S/128, B*H

  gemm_wmma<false, true, true><<<gproj, 256, 0, stream>>>(Q, Wq, bq, Qlev, qh);
  gemm_wmma<false, true, true><<<gproj, 256, 0, stream>>>(K, Wk, bk, Klev, kh);
  gemm_wmma<false, true, true><<<gproj, 256, 0, stream>>>(V, Wv, bv, Vlev, vh);
  attn_wmma<<<gattn, 256, 0, stream>>>(qh, kh, vh, ctx);
  gemm_wmma<true, false, false><<<gproj, 256, 0, stream>>>(ctx, Wo, bo, nullptr,
                                                           d_out);
}
